// SAGEReg_43130061586945
// MI455X (gfx1250) — matrix operand order repaired and checked by reading, not compiled, not run on hardware
//
#include <hip/hip_runtime.h>

// ---- WMMA vector types (CDNA5 / gfx1250, wave32) ----
typedef __attribute__((ext_vector_type(16))) _Float16 v16h;
typedef __attribute__((ext_vector_type(8)))  float    v8f;

#define SAGE_N 100000
#define SAGE_E 1600000

// ---------------- utility kernels ----------------

__global__ void sage_zero_f32(float* __restrict__ p, int n) {
    int i = blockIdx.x * blockDim.x + threadIdx.x;
    if (i < n) p[i] = 0.0f;
}

// Build concatenated, transposed f16 weight matrix:
//   WT[n*128 + k] = (k < 64) ? Wl[n][k] : Wr[n][k-64]
// n-major so each WMMA lane's B fragment (16 halves) is contiguous.
__global__ void sage_pack_weights(const float* __restrict__ Wl,
                                  const float* __restrict__ Wr,
                                  _Float16* __restrict__ WT, int nOut) {
    int t = blockIdx.x * blockDim.x + threadIdx.x;
    if (t >= nOut * 128) return;
    int n = t >> 7;
    int k = t & 127;
    float v = (k < 64) ? Wl[n * 64 + k] : Wr[n * 64 + (k - 64)];
    WT[t] = (_Float16)v;
}

__global__ void sage_degree(const int* __restrict__ dst, float* __restrict__ cnt, int E) {
    int e = blockIdx.x * blockDim.x + threadIdx.x;
    if (e < E) atomicAdd(&cnt[dst[e]], 1.0f);
}

// 4 threads per edge, 16 features each: float4 gathers from feat[src],
// L2-resident float atomics onto agg[dst]. Index traffic = E*4*8B ~ 51 MB.
__global__ void sage_scatter_add(const int* __restrict__ src, const int* __restrict__ dst,
                                 const float* __restrict__ feat, float* __restrict__ agg, int E) {
    int t = blockIdx.x * blockDim.x + threadIdx.x;
    if (t >= E * 4) return;
    int e = t >> 2;
    int c = (t & 3) << 4;                       // feature base: 0,16,32,48
    int s = src[e];
    int d = dst[e];
    const float4* xp = reinterpret_cast<const float4*>(feat + s * 64 + c);
    float* ap = agg + d * 64 + c;
#pragma unroll
    for (int q = 0; q < 4; ++q) {
        const float4 v = xp[q];
        atomicAdd(ap + q * 4 + 0, v.x);
        atomicAdd(ap + q * 4 + 1, v.y);
        atomicAdd(ap + q * 4 + 2, v.z);
        atomicAdd(ap + q * 4 + 3, v.w);
    }
}

// ---------------- fused SAGE layer via WMMA ----------------
// Computes: out = relu( (agg/deg) @ Wl^T + bias + xin @ Wr^T )
// as one K=128 GEMM against packed WT = [Wl^T ; Wr^T] (f16, staged in LDS).
// One wave -> one 16(M) x 16(N) f32 tile, 4 steps of v_wmma_f32_16x16x32_f16.
__global__ void sage_layer_wmma(const float* __restrict__ agg,
                                const float* __restrict__ cnt,
                                const float* __restrict__ xin,
                                const _Float16* __restrict__ WT,
                                const float* __restrict__ bias,
                                float* __restrict__ out,
                                int nNodes, int nOut) {
    // Stage full weight matrix (nOut*128 halves, <=16KB) into LDS.
    __shared__ __align__(32) _Float16 sWT[64 * 128];
    {
        const uint4* gsrc = reinterpret_cast<const uint4*>(WT);
        uint4*       ldst = reinterpret_cast<uint4*>(sWT);
        const int chunks = nOut * 16;           // 16B chunks
        for (int i = threadIdx.x; i < chunks; i += blockDim.x) ldst[i] = gsrc[i];
    }
    __syncthreads();

    const int wid    = (blockIdx.x * blockDim.x + threadIdx.x) >> 5;
    const int lane   = threadIdx.x & 31;
    const int ntiles = nOut >> 4;
    const int mtiles = nNodes >> 4;
    if (wid >= mtiles * ntiles) return;         // wave-uniform: EXEC all-1s for WMMA

    const int mtile = wid / ntiles;
    const int ntile = wid - mtile * ntiles;

    const int r16  = lane & 15;
    const int half = lane >> 4;

    const int m = mtile * 16 + r16;             // A-fragment row for this lane
    const int n = ntile * 16 + r16;             // B-fragment column / output column
    const float inv = 1.0f / fmaxf(cnt[m], 1.0f);

    const float* arow = agg + m * 64;
    const float* xrow = xin + m * 64;
    const _Float16* brow = sWT + (size_t)n * 128;   // lane's weight column, k-contiguous

    v8f c = {};
#pragma unroll
    for (int kc = 0; kc < 4; ++kc) {            // K = kc*32 .. kc*32+31
        v16h a;
#pragma unroll
        for (int j = 0; j < 8; ++j) {
            // A layout (matches B): VGPR j holds K = half*16 + j*2, +1
            int kbA = (half << 4) + (j << 1);
            float2 v;
            if (kc < 2) {                       // K in [0,64): mean-aggregated neighbors
                v = *reinterpret_cast<const float2*>(arow + kc * 32 + kbA);
                v.x *= inv; v.y *= inv;
            } else {                            // K in [64,128): root features
                v = *reinterpret_cast<const float2*>(xrow + (kc - 2) * 32 + kbA);
            }
            a[2 * j]     = (_Float16)v.x;
            a[2 * j + 1] = (_Float16)v.y;
        }
        // B layout: lanes 0-15 hold K=0..15, lanes 16-31 hold K=16..31 per chunk;
        // lane's 16 halves are contiguous -> 2x ds_load_b128 from LDS.
        v16h b = *reinterpret_cast<const v16h*>(brow + kc * 32 + half * 16);

        // (neg_a, A, neg_b, B, c_mod, C, reuse_a, reuse_b)
        c = __builtin_amdgcn_wmma_f32_16x16x32_f16(false, a, false, b, (short)0, c, false, false);
    }

    // C/D 16x16 f32 layout: VGPR r -> row M = r + 8*half, column N = lane&15
    const float bv = bias[n];
    const int rowbase = mtile * 16 + half * 8;
#pragma unroll
    for (int r = 0; r < 8; ++r) {
        float v = c[r] + bv;
        v = fmaxf(v, 0.0f);
        out[(size_t)(rowbase + r) * nOut + n] = v;
    }
}

// ---------------- head: out[i] = h2[i,:] . Wh + bh ----------------
__global__ void sage_head(const float* __restrict__ h2, const float* __restrict__ Wh,
                          const float* __restrict__ bh, float* __restrict__ out, int nNodes) {
    int i = blockIdx.x * blockDim.x + threadIdx.x;
    if (i >= nNodes) return;
    const float* row = h2 + (size_t)i * 32;
    float s = bh[0];
#pragma unroll
    for (int j = 0; j < 32; ++j) s += row[j] * Wh[j];
    out[i] = s;
}

// ---------------- launch ----------------

extern "C" void kernel_launch(void* const* d_in, const int* in_sizes, int n_in,
                              void* d_out, int out_size, void* d_ws, size_t ws_size,
                              hipStream_t stream) {
    const float* x   = (const float*)d_in[0];
    const int*   ei  = (const int*)d_in[1];
    const float* W1l = (const float*)d_in[2];
    const float* b1l = (const float*)d_in[3];
    const float* W1r = (const float*)d_in[4];
    const float* W2l = (const float*)d_in[5];
    const float* b2l = (const float*)d_in[6];
    const float* W2r = (const float*)d_in[7];
    const float* Wh  = (const float*)d_in[8];
    const float* bh  = (const float*)d_in[9];
    float* out = (float*)d_out;

    const int N = SAGE_N;
    const int E = SAGE_E;
    const int* srcp = ei;        // ei[0,:]
    const int* dstp = ei + E;    // ei[1,:]

    // Carve workspace (256B-aligned chunks). Total ~64.5 MB.
    char*  ws  = (char*)d_ws;
    size_t off = 0;
    auto alloc = [&](size_t bytes) -> void* {
        void* p = (void*)(ws + off);
        off += (bytes + 255) & ~(size_t)255;
        return p;
    };
    float*    agg = (float*)alloc((size_t)N * 64 * sizeof(float));  // reused for both layers
    float*    cnt = (float*)alloc((size_t)N * sizeof(float));
    float*    h1  = (float*)alloc((size_t)N * 64 * sizeof(float));
    float*    h2  = (float*)alloc((size_t)N * 32 * sizeof(float));
    _Float16* WT1 = (_Float16*)alloc((size_t)64 * 128 * sizeof(_Float16));
    _Float16* WT2 = (_Float16*)alloc((size_t)32 * 128 * sizeof(_Float16));

    const int TB = 256;
    const int nAgg = N * 64;

    // ---- prep ----
    sage_zero_f32<<<(nAgg + TB - 1) / TB, TB, 0, stream>>>(agg, nAgg);
    sage_zero_f32<<<(N + TB - 1) / TB, TB, 0, stream>>>(cnt, N);
    sage_pack_weights<<<(64 * 128 + TB - 1) / TB, TB, 0, stream>>>(W1l, W1r, WT1, 64);
    sage_pack_weights<<<(32 * 128 + TB - 1) / TB, TB, 0, stream>>>(W2l, W2r, WT2, 32);
    sage_degree<<<(E + TB - 1) / TB, TB, 0, stream>>>(dstp, cnt, E);

    // ---- layer 1 ----
    sage_scatter_add<<<(E * 4 + TB - 1) / TB, TB, 0, stream>>>(srcp, dstp, x, agg, E);
    {
        int waves = (N / 16) * (64 / 16);          // 25000 waves
        int threads = waves * 32;
        sage_layer_wmma<<<(threads + TB - 1) / TB, TB, 0, stream>>>(
            agg, cnt, x, WT1, b1l, h1, N, 64);
    }

    // ---- layer 2 (reuse agg buffer) ----
    sage_zero_f32<<<(nAgg + TB - 1) / TB, TB, 0, stream>>>(agg, nAgg);
    sage_scatter_add<<<(E * 4 + TB - 1) / TB, TB, 0, stream>>>(srcp, dstp, h1, agg, E);
    {
        int waves = (N / 16) * (32 / 16);          // 12500 waves
        int threads = waves * 32;
        sage_layer_wmma<<<(threads + TB - 1) / TB, TB, 0, stream>>>(
            agg, cnt, h1, WT2, b2l, h2, N, 32);
    }

    // ---- head ----
    sage_head<<<(N + TB - 1) / TB, TB, 0, stream>>>(h2, Wh, bh, out, N);
}
